// PointNet2Encoder_29635274342492
// MI455X (gfx1250) — matrix operand order repaired and checked by reading, not compiled, not run on hardware
//
#include <hip/hip_runtime.h>
#include <hip/hip_bf16.h>

typedef _Float16 f16;
typedef _Float16 v16h __attribute__((ext_vector_type(16)));
typedef _Float16 v8h  __attribute__((ext_vector_type(8)));
typedef float    v8f  __attribute__((ext_vector_type(8)));

union V16 { v16h v; v8h h[2]; };

static __device__ __forceinline__ v8f wmma16(const V16& a, const V16& b, v8f c) {
  return __builtin_amdgcn_wmma_f32_16x16x32_f16(
      /*neg_a=*/false, a.v, /*neg_b=*/false, b.v,
      /*c_mod=*/(short)0, c, /*reuse_a=*/false, /*reuse_b=*/false);
}

static __device__ __forceinline__ void wait_async0() {
#if __has_builtin(__builtin_amdgcn_s_wait_asynccnt)
  __builtin_amdgcn_s_wait_asynccnt(0);
#else
  asm volatile("s_wait_asynccnt 0x0" ::: "memory");
#endif
}

// Async DMA: copy 16 bytes global -> LDS without a VGPR round trip (ASYNCcnt).
static __device__ __forceinline__ void async_g2l_b128(unsigned lds_addr,
                                                      const void* gaddr) {
  asm volatile("global_load_async_to_lds_b128 %0, %1, off"
               :: "v"(lds_addr), "v"(gaddr) : "memory");
}

// ---------------------------------------------------------------- utilities

__global__ void k_split(const float* __restrict__ in, float* __restrict__ xyz,
                        float* __restrict__ nrm, int B, int N) {
  int t = blockIdx.x * 256 + threadIdx.x;
  int tot = B * N * 3;
  if (t >= tot) return;
  int b = t / (N * 3); int rem = t - b * N * 3; int n = rem / 3; int c = rem - n * 3;
  xyz[t] = in[(size_t)b * 6 * N + (size_t)c * N + n];
  nrm[t] = in[(size_t)b * 6 * N + (size_t)(c + 3) * N + n];
}

// Farthest point sampling: one block per batch, dist[] resident in LDS.
__global__ void k_fps(const float* __restrict__ xyz, int N, int npoint,
                      int* __restrict__ fidx) {
  __shared__ float dist[4096];
  __shared__ float red[256];
  __shared__ int   redi[256];
  __shared__ int   s_far;
  int b = blockIdx.x, tid = threadIdx.x;
  const float* p = xyz + (size_t)b * N * 3;
  for (int i = tid; i < N; i += 256) dist[i] = 1e10f;
  __syncthreads();
  int far = 0;
  for (int it = 0; it < npoint; ++it) {
    if (tid == 0) fidx[b * npoint + it] = far;
    float cx = p[far * 3], cy = p[far * 3 + 1], cz = p[far * 3 + 2];
    float best = -1.0f; int bi = N;
    for (int i = tid; i < N; i += 256) {
      float dx = p[i * 3] - cx, dy = p[i * 3 + 1] - cy, dz = p[i * 3 + 2] - cz;
      float d = dx * dx + dy * dy + dz * dz;
      float nd = fminf(dist[i], d);
      dist[i] = nd;
      if (nd > best || (nd == best && i < bi)) { best = nd; bi = i; }
    }
    red[tid] = best; redi[tid] = bi;
    __syncthreads();
    for (int s = 128; s > 0; s >>= 1) {
      if (tid < s) {
        if (red[tid + s] > red[tid] ||
            (red[tid + s] == red[tid] && redi[tid + s] < redi[tid])) {
          red[tid] = red[tid + s]; redi[tid] = redi[tid + s];
        }
      }
      __syncthreads();
    }
    if (tid == 0) s_far = redi[0];
    __syncthreads();
    far = s_far;
    __syncthreads();
  }
}

__global__ void k_gather(const float* __restrict__ xyz, const int* __restrict__ fidx,
                         float* __restrict__ out, int B, int N, int np) {
  int t = blockIdx.x * 256 + threadIdx.x;
  if (t >= B * np * 3) return;
  int b = t / (np * 3); int rem = t - b * np * 3; int s = rem / 3; int d = rem - s * 3;
  out[t] = xyz[(size_t)b * N * 3 + (size_t)fidx[b * np + s] * 3 + d];
}

// Ball query: one wave32 per centroid; ballot + prefix popcount gives the
// first-K ascending in-radius indices, padded with the first hit.
__global__ void k_ball(const float* __restrict__ xyz, const float* __restrict__ nxyz,
                       int N, int np, int K, float r2, int* __restrict__ gidx) {
  int wave = threadIdx.x >> 5, lane = threadIdx.x & 31;
  int s = blockIdx.x * 8 + wave;
  if (s >= np) return;
  float cx = nxyz[s * 3], cy = nxyz[s * 3 + 1], cz = nxyz[s * 3 + 2];
  int* out = gidx + (size_t)s * K;
  int cnt = 0, firstid = 0; bool have = false;
  for (int base = 0; base < N && cnt < K; base += 32) {
    int i = base + lane;
    float dx = xyz[i * 3] - cx, dy = xyz[i * 3 + 1] - cy, dz = xyz[i * 3 + 2] - cz;
    bool inr = (dx * dx + dy * dy + dz * dz) <= r2;
    unsigned mask = (unsigned)__ballot(inr);
    if (!have && mask) { firstid = base + __ffs((int)mask) - 1; have = true; }
    if (inr) {
      int pos = cnt + __popc(mask & ((1u << lane) - 1u));
      if (pos < K) out[pos] = i;
    }
    cnt += __popc(mask);
  }
  for (int pidx = cnt + lane; pidx < K; pidx += 32) out[pidx] = firstid;
}

// Build the f16 A-matrix: [points-features | (xyz - center) | zero pad]
__global__ void k_group(const float* __restrict__ xyz, const float* __restrict__ nxyz,
                        const float* __restrict__ pf32, const f16* __restrict__ pf16,
                        int C, const int* __restrict__ gidx, int K, int cinp,
                        f16* __restrict__ Aout, int rows) {
  int r = blockIdx.x * 256 + threadIdx.x;
  if (r >= rows) return;
  int s = r / K;
  int id = gidx[r];
  f16* dst = Aout + (size_t)r * cinp;
  if (pf16) {
    const f16* q = pf16 + (size_t)id * C;
    for (int c = 0; c < C; ++c) dst[c] = q[c];
  } else {
    const float* q = pf32 + (size_t)id * C;
    for (int c = 0; c < C; ++c) dst[c] = (f16)q[c];
  }
#pragma unroll
  for (int d = 0; d < 3; ++d) dst[C + d] = (f16)(xyz[id * 3 + d] - nxyz[s * 3 + d]);
  for (int c = C + 3; c < cinp; ++c) dst[c] = (f16)0.f;
}

// group-all for SA3: [xyz | points | pad] (note reversed concat order)
__global__ void k_group_all(const float* __restrict__ nx2, const f16* __restrict__ l2p,
                            f16* __restrict__ Aout, int B) {
  int r = blockIdx.x * 256 + threadIdx.x;
  if (r >= B * 128) return;
  f16* dst = Aout + (size_t)r * 672;
#pragma unroll
  for (int d = 0; d < 3; ++d) dst[d] = (f16)nx2[r * 3 + d];
  const f16* q = l2p + (size_t)r * 640;
  for (int c = 0; c < 640; ++c) dst[3 + c] = q[c];
  for (int c = 643; c < 672; ++c) dst[c] = (f16)0.f;
}

__global__ void k_pack_w(const float* __restrict__ W, f16* __restrict__ Wp,
                         int cin, int cinp, int cout) {
  int t = blockIdx.x * 256 + threadIdx.x;
  if (t >= cinp * cout) return;
  int i = t / cout, j = t - i * cout;
  Wp[t] = (i < cin) ? (f16)W[(size_t)i * cout + j] : (f16)0.f;
}

// Fold bias + BN affine into per-channel scale/shift: relu(acc*s + t)
__global__ void k_affine(const float* __restrict__ b, const float* __restrict__ g,
                         const float* __restrict__ bt, float* __restrict__ sc,
                         float* __restrict__ sh, int n) {
  int c = blockIdx.x * 64 + threadIdx.x;
  if (c >= n) return;
  float inv = rsqrtf(1.0f + 1e-5f);
  float s = g[c] * inv;
  sc[c] = s;
  sh[c] = b[c] * s + bt[c];
}

// ------------------------------------------------------- WMMA GEMM (generic)
// 32x32 block tile, 4 waves, one 16x16 WMMA tile per wave. Used for the few
// layers with cout % 64 != 0 (cout = 32, 96).
__global__ void __launch_bounds__(128)
k_gemm32(const f16* __restrict__ A, const f16* __restrict__ W,
         const float* __restrict__ sc, const float* __restrict__ sh,
         f16* __restrict__ C, int rows, int cinp, int cout) {
  __shared__ f16 As[32][40];   // [m][k]   (40-half stride = 80B, 16B aligned)
  __shared__ f16 Bt[32][40];   // [n][k]
  int t = threadIdx.x;
  int row0 = blockIdx.x * 32, col0 = blockIdx.y * 32;
  int wave = t >> 5, lane = t & 31;
  int rt = wave >> 1, ct = wave & 1;
  int m = (lane & 15) + rt * 16;
  int h = (lane >> 4) * 8;
  int n = (lane & 15) + ct * 16;
  int kb16 = (lane >> 4) * 16;
  int ar = t >> 2, ac8 = (t & 3) * 8;
  int bk = t >> 2, bn8 = (t & 3) * 8;
  const f16* Arow = A + (size_t)(row0 + ar) * cinp + ac8;
  v8f acc = {};
  for (int kb = 0; kb < cinp; kb += 32) {
    *(v8h*)&As[ar][ac8] = *(const v8h*)(Arow + kb);
    v8h w = *(const v8h*)&W[(size_t)(kb + bk) * cout + col0 + bn8];
#pragma unroll
    for (int j = 0; j < 8; ++j) Bt[bn8 + j][bk] = w[j];
    __syncthreads();
    V16 a, bf;
    a.h[0]  = *(const v8h*)&As[m][h];
    a.h[1]  = *(const v8h*)&As[m][16 + h];
    bf.h[0] = *(const v8h*)&Bt[n][h];
    bf.h[1] = *(const v8h*)&Bt[n][16 + h];
    acc = wmma16(a, bf, acc);
    __syncthreads();
  }
  int col = col0 + ct * 16 + (lane & 15);
  float s = sc[col], b = sh[col];
  int rbase = row0 + rt * 16 + (lane >> 4) * 8;
#pragma unroll
  for (int r = 0; r < 8; ++r) {
    float v = fmaxf(acc[r] * s + b, 0.f);
    C[(size_t)(rbase + r) * cout + col] = (f16)v;
  }
}

// --------------------------------------------- WMMA GEMM (register-blocked)
// 64x64 block tile, 4 waves, each wave owns a 2x2 grid of 16x16 accumulators:
// 4 WMMAs per wave per K-step with 2x fragment reuse. A tile is staged with
// GLOBAL_LOAD_ASYNC_TO_LDS_B128 (ASYNCcnt); B is transposed through VGPRs
// into the WMMA B-fragment layout.
__global__ void __launch_bounds__(128)
k_gemm64(const f16* __restrict__ A, const f16* __restrict__ W,
         const float* __restrict__ sc, const float* __restrict__ sh,
         f16* __restrict__ C, int rows, int cinp, int cout) {
  __shared__ f16 As[64][40];   // [m][k]
  __shared__ f16 Bt[64][40];   // [n][k]
  int t = threadIdx.x;
  int row0 = blockIdx.x * 64, col0 = blockIdx.y * 64;
  int wave = t >> 5, lane = t & 31;
  int wr = wave >> 1, wc = wave & 1;
  int lrow = lane & 15;
  int h = (lane >> 4) * 8;
  int kb16 = (lane >> 4) * 16;
  // staging coords: A = 64 rows x 32 halves (async), B = 32 k x 64 n
  int ar = t >> 1, ac16 = (t & 1) * 16;
  int bk = t >> 2, bn16 = (t & 3) * 16;
  const f16* Ag = A + (size_t)(row0 + ar) * cinp + ac16;
  unsigned ldsA0 = (unsigned)(size_t)&As[ar][ac16];
  unsigned ldsA1 = ldsA0 + 16;
  v8f acc00 = {}, acc01 = {}, acc10 = {}, acc11 = {};
  for (int kb = 0; kb < cinp; kb += 32) {
    async_g2l_b128(ldsA0, Ag + kb);
    async_g2l_b128(ldsA1, Ag + kb + 8);
    const f16* Wg = &W[(size_t)(kb + bk) * cout + col0 + bn16];
    v8h w0 = *(const v8h*)Wg;
    v8h w1 = *(const v8h*)(Wg + 8);
#pragma unroll
    for (int j = 0; j < 8; ++j) Bt[bn16 + j][bk] = w0[j];
#pragma unroll
    for (int j = 0; j < 8; ++j) Bt[bn16 + 8 + j][bk] = w1[j];
    wait_async0();
    __syncthreads();
    V16 a0, a1, b0, b1;
    int m0 = wr * 32 + lrow, n0 = wc * 32 + lrow;
    a0.h[0] = *(const v8h*)&As[m0][h];
    a0.h[1] = *(const v8h*)&As[m0][16 + h];
    a1.h[0] = *(const v8h*)&As[m0 + 16][h];
    a1.h[1] = *(const v8h*)&As[m0 + 16][16 + h];
    b0.h[0] = *(const v8h*)&Bt[n0][h];
    b0.h[1] = *(const v8h*)&Bt[n0][16 + h];
    b1.h[0] = *(const v8h*)&Bt[n0 + 16][h];
    b1.h[1] = *(const v8h*)&Bt[n0 + 16][16 + h];
    acc00 = wmma16(a0, b0, acc00);
    acc01 = wmma16(a0, b1, acc01);
    acc10 = wmma16(a1, b0, acc10);
    acc11 = wmma16(a1, b1, acc11);
    __syncthreads();
  }
  int c0 = col0 + wc * 32 + lrow, c1 = c0 + 16;
  float s0 = sc[c0], t0 = sh[c0], s1 = sc[c1], t1 = sh[c1];
  int r0 = row0 + wr * 32 + (lane >> 4) * 8, r1 = r0 + 16;
#pragma unroll
  for (int r = 0; r < 8; ++r) {
    C[(size_t)(r0 + r) * cout + c0] = (f16)fmaxf(acc00[r] * s0 + t0, 0.f);
    C[(size_t)(r0 + r) * cout + c1] = (f16)fmaxf(acc01[r] * s1 + t1, 0.f);
    C[(size_t)(r1 + r) * cout + c0] = (f16)fmaxf(acc10[r] * s0 + t0, 0.f);
    C[(size_t)(r1 + r) * cout + c1] = (f16)fmaxf(acc11[r] * s1 + t1, 0.f);
  }
}

// ---------------------------------------------------------------- max pools

__global__ void k_maxpool_h(const f16* __restrict__ F, f16* __restrict__ dst,
                            int np, int K, int cout, int Ctot, int coff) {
  int t = blockIdx.x * 256 + threadIdx.x;
  if (t >= np * cout) return;
  int s = t / cout, c = t - s * cout;
  float m = -1e30f;
  const f16* p = F + (size_t)s * K * cout + c;
  for (int k = 0; k < K; ++k) m = fmaxf(m, (float)p[(size_t)k * cout]);
  dst[(size_t)s * Ctot + coff + c] = (f16)m;
}

__global__ void k_maxpool_f(const f16* __restrict__ F, float* __restrict__ out,
                            int B, int P, int Cout) {
  int t = blockIdx.x * 256 + threadIdx.x;
  if (t >= B * Cout) return;
  int b = t / Cout, c = t - b * Cout;
  float m = -1e30f;
  const f16* p = F + (size_t)b * P * Cout + c;
  for (int k = 0; k < P; ++k) m = fmaxf(m, (float)p[(size_t)k * Cout]);
  out[t] = m;
}

// ---------------------------------------------------------------- host side

extern "C" void kernel_launch(void* const* d_in, const int* in_sizes, int n_in,
                              void* d_out, int out_size, void* d_ws, size_t ws_size,
                              hipStream_t stream) {
  const int B = 16, N1 = 4096, NP1 = 512, NP2 = 128;
  char* ws = (char*)d_ws;
  size_t off = 0;
  auto alloc = [&](size_t bytes) -> char* {
    char* p = ws + off; off = (off + bytes + 255) & ~(size_t)255; return p;
  };
  float* xyz   = (float*)alloc((size_t)B * N1 * 3 * 4);
  float* nrm   = (float*)alloc((size_t)B * N1 * 3 * 4);
  int*   fidx1 = (int*)  alloc((size_t)B * NP1 * 4);
  float* nx1   = (float*)alloc((size_t)B * NP1 * 3 * 4);
  f16*   l1p   = (f16*)  alloc((size_t)B * NP1 * 320 * 2);
  int*   fidx2 = (int*)  alloc((size_t)B * NP2 * 4);
  float* nx2   = (float*)alloc((size_t)B * NP2 * 3 * 4);
  f16*   l2p   = (f16*)  alloc((size_t)B * NP2 * 640 * 2);
  int*   gidx  = (int*)  alloc((size_t)NP1 * 128 * 4);
  f16*   bufA  = (f16*)  alloc((size_t)16384 * 352 * 2);   // max A tile (per batch)
  f16*   bufB  = (f16*)  alloc((size_t)65536 * 128 * 2);   // layer ping
  f16*   bufC  = (f16*)  alloc((size_t)65536 * 128 * 2);   // layer pong

  struct LD { int cin, cout, cinp; const float *W, *bb, *g, *bt;
              f16* Wp; float* sc; float* sh; };
  LD L[21]; int nL = 0; int argi = 1;
  auto addBlock = [&](int cin, const int* mlp, int nl) {
    int c = cin;
    for (int i = 0; i < nl; ++i) {
      LD& x = L[nL++];
      x.cin = c; x.cout = mlp[i]; x.cinp = (c + 31) & ~31;
      x.W  = (const float*)d_in[argi++]; x.bb = (const float*)d_in[argi++];
      x.g  = (const float*)d_in[argi++]; x.bt = (const float*)d_in[argi++];
      x.Wp = (f16*)alloc((size_t)x.cinp * x.cout * 2);
      x.sc = (float*)alloc((size_t)x.cout * 4);
      x.sh = (float*)alloc((size_t)x.cout * 4);
      c = mlp[i];
    }
  };
  const int m1[3][3] = {{32, 32, 64}, {64, 64, 128}, {64, 96, 128}};
  const int m2[3][3] = {{64, 64, 128}, {128, 128, 256}, {128, 128, 256}};
  const int m3[3] = {256, 512, 1024};
  for (int s = 0; s < 3; ++s) addBlock(6,   m1[s], 3);
  for (int s = 0; s < 3; ++s) addBlock(323, m2[s], 3);
  addBlock(643, m3, 3);

  for (int i = 0; i < nL; ++i) {
    int tot = L[i].cinp * L[i].cout;
    k_pack_w<<<dim3((tot + 255) / 256), dim3(256), 0, stream>>>(
        L[i].W, L[i].Wp, L[i].cin, L[i].cinp, L[i].cout);
    k_affine<<<dim3((L[i].cout + 63) / 64), dim3(64), 0, stream>>>(
        L[i].bb, L[i].g, L[i].bt, L[i].sc, L[i].sh, L[i].cout);
  }

  { int tot = B * N1 * 3;
    k_split<<<dim3((tot + 255) / 256), dim3(256), 0, stream>>>(
        (const float*)d_in[0], xyz, nrm, B, N1); }

  auto runMLP = [&](int l0, int rows) {
    const f16* src = bufA; f16* dsts[3] = {bufB, bufC, bufB};
    for (int i = 0; i < 3; ++i) {
      const LD& x = L[l0 + i];
      if ((x.cout & 63) == 0 && (rows & 63) == 0) {
        k_gemm64<<<dim3(rows / 64, x.cout / 64), dim3(128), 0, stream>>>(
            src, x.Wp, x.sc, x.sh, dsts[i], rows, x.cinp, x.cout);
      } else {
        k_gemm32<<<dim3(rows / 32, x.cout / 32), dim3(128), 0, stream>>>(
            src, x.Wp, x.sc, x.sh, dsts[i], rows, x.cinp, x.cout);
      }
      src = dsts[i];
    }
  };

  // ---- SA1 ----
  k_fps<<<dim3(B), dim3(256), 0, stream>>>(xyz, N1, NP1, fidx1);
  { int tot = B * NP1 * 3;
    k_gather<<<dim3((tot + 255) / 256), dim3(256), 0, stream>>>(
        xyz, fidx1, nx1, B, N1, NP1); }

  const int   K1[3] = {16, 32, 128};  const float R1[3] = {0.1f, 0.2f, 0.4f};
  const int   K2[3] = {32, 64, 128};  const float R2[3] = {0.2f, 0.4f, 0.8f};
  const int   coff1[3] = {0, 64, 192}, coff2[3] = {0, 128, 384};

  for (int b = 0; b < B; ++b) {
    for (int s = 0; s < 3; ++s) {
      int K = K1[s]; int rows = NP1 * K;
      k_ball<<<dim3(NP1 / 8), dim3(256), 0, stream>>>(
          xyz + (size_t)b * N1 * 3, nx1 + (size_t)b * NP1 * 3,
          N1, NP1, K, R1[s] * R1[s], gidx);
      k_group<<<dim3((rows + 255) / 256), dim3(256), 0, stream>>>(
          xyz + (size_t)b * N1 * 3, nx1 + (size_t)b * NP1 * 3,
          nrm + (size_t)b * N1 * 3, (const f16*)nullptr, 3, gidx, K, 32, bufA, rows);
      runMLP(s * 3, rows);
      int cout = L[s * 3 + 2].cout;
      k_maxpool_h<<<dim3((NP1 * cout + 255) / 256), dim3(256), 0, stream>>>(
          bufB, l1p + (size_t)b * NP1 * 320, NP1, K, cout, 320, coff1[s]);
    }
  }

  // ---- SA2 ----
  k_fps<<<dim3(B), dim3(256), 0, stream>>>(nx1, NP1, NP2, fidx2);
  { int tot = B * NP2 * 3;
    k_gather<<<dim3((tot + 255) / 256), dim3(256), 0, stream>>>(
        nx1, fidx2, nx2, B, NP1, NP2); }

  for (int b = 0; b < B; ++b) {
    for (int s = 0; s < 3; ++s) {
      int K = K2[s]; int rows = NP2 * K;
      k_ball<<<dim3(NP2 / 8), dim3(256), 0, stream>>>(
          nx1 + (size_t)b * NP1 * 3, nx2 + (size_t)b * NP2 * 3,
          NP1, NP2, K, R2[s] * R2[s], gidx);
      k_group<<<dim3((rows + 255) / 256), dim3(256), 0, stream>>>(
          nx1 + (size_t)b * NP1 * 3, nx2 + (size_t)b * NP2 * 3,
          (const float*)nullptr, l1p + (size_t)b * NP1 * 320,
          320, gidx, K, 352, bufA, rows);
      runMLP(9 + s * 3, rows);
      int cout = L[9 + s * 3 + 2].cout;
      k_maxpool_h<<<dim3((NP2 * cout + 255) / 256), dim3(256), 0, stream>>>(
          bufB, l2p + (size_t)b * NP2 * 640, NP2, K, cout, 640, coff2[s]);
    }
  }

  // ---- SA3 (group all) ----
  { int rows = B * NP2;
    k_group_all<<<dim3((rows + 255) / 256), dim3(256), 0, stream>>>(
        nx2, l2p, bufA, B);
    runMLP(18, rows);
    k_maxpool_f<<<dim3((B * 1024 + 255) / 256), dim3(256), 0, stream>>>(
        bufB, (float*)d_out, B, NP2, 1024); }

  (void)in_sizes; (void)n_in; (void)out_size; (void)ws_size;
}
